// GCN_CharNet_70832600646034
// MI455X (gfx1250) — matrix operand order repaired and checked by reading, not compiled, not run on hardware
//
#include <hip/hip_runtime.h>
#include <hip/hip_bf16.h>
#include <math.h>

// ---------------- Problem constants ----------------
#define B_   16
#define P_   5
#define S_   200
#define L_   81
#define BF_  32
#define OUT_ 64
#define HB_  162
#define HO_  32
#define HM_  162
#define BPS_ (B_*P_*S_)   // 16000
#define SK_  8            // split-K factor for final projection

typedef __attribute__((ext_vector_type(16))) _Float16 v16h;
typedef __attribute__((ext_vector_type(8)))  float    v8f;

#define WMMA_F32(a,b,c) __builtin_amdgcn_wmma_f32_16x16x32_f16(false,(a),false,(b),(short)0,(c),false,false)

// ---- Fragment loaders per CDNA5 wave32 WMMA VGPR layouts (CDNA5 ISA) ----
// A (16x32 f16): lane L holds M = L%16; K slots: lanes 0-15 -> {0..7, 16..23},
// lanes 16-31 -> {8..15, 24..31}. Each 8-slot run is 16B contiguous -> b128 loads.
__device__ inline v16h frag_a_h(const _Float16* A, int ld, int mBase, int kBase) {
  int lane = (int)(threadIdx.x & 31);
  int m  = mBase + (lane & 15);
  int kb = kBase + 8 * (lane >> 4);
  v16h a;
#pragma unroll
  for (int j = 0; j < 16; ++j) {
    int k = (j < 8) ? (kb + j) : (kb + 16 + (j - 8));
    a[j] = A[m * ld + k];
  }
  return a;
}
__device__ inline v16h frag_a_f32(const float* A, int ld, int mBase, int kBase) {
  int lane = (int)(threadIdx.x & 31);
  int m  = mBase + (lane & 15);
  int kb = kBase + 8 * (lane >> 4);
  v16h a;
#pragma unroll
  for (int j = 0; j < 16; ++j) {
    int k = (j < 8) ? (kb + j) : (kb + 16 + (j - 8));
    a[j] = (_Float16)A[m * ld + k];
  }
  return a;
}
// B fragment, PACKED layout: tile(kt,nt) stores lane L's 16 values contiguously:
// elem j = B[kt*32 + 8*(L>>4) + (j<8 ? j : 8+j)][nt*16 + (L&15)]  -> one aligned 32B v16h load.
__device__ inline v16h frag_b_pk(const _Float16* Bp, int NTt, int kt, int nt) {
  int lane = (int)(threadIdx.x & 31);
  const _Float16* p = Bp + ((((size_t)kt * NTt + nt) << 5) + lane) * 16;
  return *(const v16h*)p;
}
// packed-store address for writing a B-operand value at (k,n), NTt n-tiles
__device__ inline int bpk_idx(int k, int n, int NTt) {
  int kt = k >> 5, k32 = k & 31;
  int Lp = (((k32 >> 3) & 1) << 4) | (n & 15);
  return (((kt * NTt + (n >> 4)) << 5) + Lp) * 16 + ((k32 & 7) | ((k32 >> 4) << 3));
}

__device__ inline float sigmoidf_(float x) { return 1.0f / (1.0f + expf(-x)); }

// ---- weight convert: W[N][K] f32 -> fragment-packed f16 B operand [Kpad x Npad] ----
__global__ void k_conv_wT_pk(const float* w, _Float16* dst, int N, int K, int Kpad, int Npad) {
  int NTt = Npad >> 4;
  int total = Kpad * Npad;   // multiple of 512
  for (int t = blockIdx.x * blockDim.x + threadIdx.x; t < total; t += gridDim.x * blockDim.x) {
    int tile = t >> 9;
    int rem  = t & 511;
    int L = rem >> 4, j = rem & 15;
    int kt = tile / NTt, nt = tile % NTt;
    int k = kt * 32 + ((L >> 4) << 3) + (j < 8 ? j : 8 + j);
    int n = nt * 16 + (L & 15);
    float v = (k < K && n < N) ? w[n * K + k] : 0.0f;
    dst[t] = (_Float16)v;
  }
}
// adj [81][81] f32 -> [96][96] f16 zero-padded (A operand, row-major)
__global__ void k_conv_adj(const float* adj, _Float16* dst) {
  for (int t = blockIdx.x * blockDim.x + threadIdx.x; t < 96 * 96; t += gridDim.x * blockDim.x) {
    int m = t / 96, k = t % 96;
    dst[t] = (m < L_ && k < L_) ? (_Float16)adj[m * L_ + k] : (_Float16)0.0f;
  }
}

// ---------------- GCN: bf[bps][162] ----------------
__global__ void k_gcn(const float* board, const float* adjF,
                      const _Float16* g1wT, const float* g1b,
                      const float* g2w, const float* g2b,
                      const _Float16* adjA, float* bf) {
  __shared__ __attribute__((aligned(32))) _Float16 sX[96 * 32];   // board tile, reused as H1
  __shared__ __attribute__((aligned(32))) _Float16 sYp[96 * 32];  // Y, B-fragment-packed (NTt=2)
  __shared__ float sT2[96 * 2];
  int bps = blockIdx.x;
  const float* bsrc = board + (size_t)bps * (L_ * BF_);
  int tid = threadIdx.x;
  for (int t = tid; t < 96 * 32; t += blockDim.x) {
    int l = t / 32, f = t % 32;
    sX[t] = (l < L_) ? (_Float16)bsrc[l * BF_ + f] : (_Float16)0.0f;
  }
  __syncthreads();
  int wave = tid >> 5, lane = tid & 31;
  // layer1: Y = X @ g1w^T + g1b  (M=96 pad, N=32, K=32); write Y packed as B operand
  for (int job = wave; job < 12; job += 4) {
    int mt = job >> 1, nt = job & 1;
    v16h a = frag_a_h(sX, 32, mt * 16, 0);
    v16h b = frag_b_pk(g1wT, 2, 0, nt);
    v8f c = {};
    c = WMMA_F32(a, b, c);
    int n = nt * 16 + (lane & 15);
    float bias = g1b[n];
#pragma unroll
    for (int r = 0; r < 8; ++r) {
      int m = mt * 16 + r + 8 * (lane >> 4);   // k-dim of next GEMM
      sYp[bpk_idx(m, n, 2)] = (_Float16)(c[r] + bias);
    }
  }
  __syncthreads();
  // H1 = relu(adj @ Y)  (K=96 pad: adjA cols>=81 are zero)
  for (int job = wave; job < 12; job += 4) {
    int mt = job >> 1, nt = job & 1;
    v8f c = {};
    for (int ks = 0; ks < 3; ++ks) {
      v16h a = frag_a_h(adjA, 96, mt * 16, ks * 32);
      v16h b = frag_b_pk(sYp, 2, ks, nt);
      c = WMMA_F32(a, b, c);
    }
#pragma unroll
    for (int r = 0; r < 8; ++r) {
      int m = mt * 16 + r + 8 * (lane >> 4);
      int n = nt * 16 + (lane & 15);
      float v = c[r];
      sX[m * 32 + n] = (_Float16)(v > 0.0f ? v : 0.0f);
    }
  }
  __syncthreads();
  // T2 = H1 @ g2w^T + g2b  (N=2, VALU)
  for (int t = tid; t < L_ * 2; t += blockDim.x) {
    int l = t >> 1, cc = t & 1;
    float acc = g2b[cc];
    for (int j = 0; j < 32; ++j) acc += (float)sX[l * 32 + j] * g2w[cc * 32 + j];
    sT2[l * 2 + cc] = acc;
  }
  __syncthreads();
  // H2 = relu(adj @ T2) -> bf
  float* dst = bf + (size_t)bps * (2 * L_);
  for (int t = tid; t < L_ * 2; t += blockDim.x) {
    int l = t >> 1, cc = t & 1;
    float acc = 0.0f;
    for (int m = 0; m < L_; ++m) acc += adjF[l * L_ + m] * sT2[m * 2 + cc];
    dst[l * 2 + cc] = acc > 0.0f ? acc : 0.0f;
  }
}

// ---------------- order features: of[bps][32] ----------------
__global__ void k_of(const float* order, const _Float16* fo1wT, const float* fo1b,
                     const _Float16* fo2wT, const float* fo2b, float* ofOut) {
  __shared__ __attribute__((aligned(32))) _Float16 st1[16 * 128];
  int rowBase = blockIdx.x * 16;
  int lane = (int)(threadIdx.x & 31);
  v16h a = frag_a_f32(order, 32, rowBase, 0);
  for (int nt = 0; nt < 8; ++nt) {
    v16h b = frag_b_pk(fo1wT, 8, 0, nt);
    v8f c = {};
    c = WMMA_F32(a, b, c);
    float bias = fo1b[nt * 16 + (lane & 15)];
#pragma unroll
    for (int r = 0; r < 8; ++r)
      st1[(r + 8 * (lane >> 4)) * 128 + nt * 16 + (lane & 15)] = (_Float16)(c[r] + bias);
  }
  __syncthreads();
  for (int nt = 0; nt < 2; ++nt) {
    v8f c = {};
    for (int ks = 0; ks < 4; ++ks) {
      v16h a2 = frag_a_h(st1, 128, 0, ks * 32);
      v16h b2 = frag_b_pk(fo2wT, 2, ks, nt);
      c = WMMA_F32(a2, b2, c);
    }
    float bias = fo2b[nt * 16 + (lane & 15)];
#pragma unroll
    for (int r = 0; r < 8; ++r)
      ofOut[(size_t)(rowBase + r + 8 * (lane >> 4)) * 32 + nt * 16 + (lane & 15)] = c[r] + bias;
  }
}

// ---------------- msg features: mf[bps][162] ----------------
__global__ void k_mf(const float* msg, const _Float16* fm1wT, const float* fm1b,
                     const _Float16* fm2wT, const float* fm2b, float* mfOut) {
  __shared__ __attribute__((aligned(32))) _Float16 sA[16 * 320];
  __shared__ __attribute__((aligned(32))) _Float16 st1[16 * 128];
  int rowBase = blockIdx.x * 16;
  int tid = threadIdx.x;
  for (int t = tid; t < 16 * 320; t += blockDim.x) {
    int r = t / 320, k = t % 320;
    sA[t] = (_Float16)msg[(size_t)(rowBase + r) * 320 + k];
  }
  __syncthreads();
  int wave = tid >> 5, lane = tid & 31;
  for (int nt = wave; nt < 8; nt += 4) {
    v8f c = {};
    for (int ks = 0; ks < 10; ++ks) {
      v16h a = frag_a_h(sA, 320, 0, ks * 32);
      v16h b = frag_b_pk(fm1wT, 8, ks, nt);
      c = WMMA_F32(a, b, c);
    }
    float bias = fm1b[nt * 16 + (lane & 15)];
#pragma unroll
    for (int r = 0; r < 8; ++r)
      st1[(r + 8 * (lane >> 4)) * 128 + nt * 16 + (lane & 15)] = (_Float16)(c[r] + bias);
  }
  __syncthreads();
  for (int nt = wave; nt < 11; nt += 4) {   // N padded 162 -> 176
    v8f c = {};
    for (int ks = 0; ks < 4; ++ks) {
      v16h a = frag_a_h(st1, 128, 0, ks * 32);
      v16h b = frag_b_pk(fm2wT, 11, ks, nt);
      c = WMMA_F32(a, b, c);
    }
    int n = nt * 16 + (lane & 15);
    if (n < HM_) {
      float bias = fm2b[n];
#pragma unroll
      for (int r = 0; r < 8; ++r)
        mfOut[(size_t)(rowBase + r + 8 * (lane >> 4)) * HM_ + n] = c[r] + bias;
    }
  }
}

// ---- Gx (transposed): GxT[blk][n][b] = x[p,s,b,:] @ wih^T + bih, blk = p*S+s ----
__global__ void k_gx(const float* feat, int F, int Kpad,
                     const _Float16* wT, const float* bih, int G, int NT,
                     float* GxT) {
  __shared__ __attribute__((aligned(32))) _Float16 sX[16 * 192];
  int blk = blockIdx.x;             // p*S + s
  int p = blk / S_, s = blk % S_;
  int tid = threadIdx.x;
  for (int t = tid; t < 16 * Kpad; t += blockDim.x) {
    int b = t / Kpad, k = t % Kpad;
    float v = 0.0f;
    if (k < F) {
      int q = s * B_ + b;                     // raw-reshape reinterpret
      int bsrc = q / S_, ssrc = q % S_;
      int bps = (bsrc * P_ + p) * S_ + ssrc;
      v = feat[(size_t)bps * F + k];
    }
    sX[t] = (_Float16)v;
  }
  __syncthreads();
  int wave = tid >> 5, lane = tid & 31;
  int KT = Kpad / 32;
  int Npad = NT * 16;
  for (int nt = wave; nt < NT; nt += 4) {
    v8f c = {};
    for (int ks = 0; ks < KT; ++ks) {
      v16h a = frag_a_h(sX, Kpad, 0, ks * 32);
      v16h b = frag_b_pk(wT, NT, ks, nt);
      c = WMMA_F32(a, b, c);
    }
    int n = nt * 16 + (lane & 15);
    float bias = (n < G) ? bih[n] : 0.0f;
    float* dst = GxT + ((size_t)blk * Npad + n) * 16;
#pragma unroll
    for (int r = 0; r < 8; ++r)
      dst[r + 8 * (lane >> 4)] = c[r] + bias;   // contiguous 32B per lane
  }
}

// ---------------- LSTM recurrence: persistent single workgroup, h/c/gates in LDS ----------------
__global__ void k_rec(const float* GxT, const _Float16* whhT, const float* bhh,
                      int H, int Kpad, int NT,
                      _Float16* hsA, int hsStride) {
  extern __shared__ char smem[];
  int Npad = NT * 16;
  _Float16* sH    = (_Float16*)smem;                               // [80][Kpad] f16 (A operand)
  float*    sGate = (float*)(smem + (size_t)80 * Kpad * sizeof(_Float16)); // [Npad][80] (transposed)
  float*    sC    = sGate + (size_t)Npad * 80;                     // [H][80] f32
  int tid = threadIdx.x, wave = tid >> 5, lane = tid & 31;
  for (int t = tid; t < 80 * Kpad; t += blockDim.x) sH[t] = (_Float16)0.0f;
  for (int t = tid; t < 80 * H; t += blockDim.x) sC[t] = 0.0f;
  __syncthreads();
  int KT = Kpad / 32;
  int G = 4 * H;
  for (int s = 0; s < S_; ++s) {
    // prefetch next step's gate inputs while this step's GEMM runs
    if (s + 1 < S_) {
      int bytes = 16 * Npad * 4;
      for (int p5 = 0; p5 < 5; ++p5) {
        const char* base = (const char*)(GxT + (size_t)(p5 * S_ + (s + 1)) * Npad * 16);
        for (int o = tid * 128; o < bytes; o += (int)blockDim.x * 128)
          __builtin_prefetch(base + o, 0, 1);
      }
    }
    for (int job = wave; job < 5 * NT; job += 32) {
      int mt = job % 5, nt = job / 5;       // mt == p
      int n = nt * 16 + (lane & 15);
      float bias = (n < G) ? bhh[n] : 0.0f;
      const float* gsrc = GxT + ((size_t)(mt * S_ + s) * Npad + n) * 16;
      v8f c;
#pragma unroll
      for (int r = 0; r < 8; ++r)
        c[r] = gsrc[r + 8 * (lane >> 4)] + bias;   // contiguous 32B per lane
      for (int ks = 0; ks < KT; ++ks) {
        v16h a  = frag_a_h(sH, Kpad, mt * 16, ks * 32);
        v16h bb = frag_b_pk(whhT, NT, ks, nt);
        c = WMMA_F32(a, bb, c);
      }
      float* gdst = sGate + (size_t)n * 80 + mt * 16;
#pragma unroll
      for (int r = 0; r < 8; ++r)
        gdst[r + 8 * (lane >> 4)] = c[r];          // contiguous 32B per lane
    }
    __syncthreads();
    for (int idx = tid; idx < 80 * H; idx += blockDim.x) {
      int k = idx / 80, r = idx % 80;              // consecutive threads -> consecutive banks
      float gi = sGate[(size_t)k * 80 + r];
      float gf = sGate[(size_t)(H + k) * 80 + r];
      float gg = sGate[(size_t)(2 * H + k) * 80 + r];
      float go = sGate[(size_t)(3 * H + k) * 80 + r];
      float cn = sigmoidf_(gf) * sC[idx] + sigmoidf_(gi) * tanhf(gg);
      float h  = sigmoidf_(go) * tanhf(cn);
      sC[idx] = cn;
      sH[r * Kpad + k] = (_Float16)h;
      hsA[(size_t)r * hsStride + s * H + k] = (_Float16)h;   // row = p*16+b
    }
    __syncthreads();
  }
}

// ---- final projection, split-K: part[(kc*20 + mt*4+nt)][m16*16+n16] (no atomics) ----
__global__ void k_proj(const _Float16* hsA, int hsStride, int KT,
                       const _Float16* fcwT, float* part) {
  int kc = blockIdx.x / 20;
  int tile = blockIdx.x % 20;
  int mt = tile >> 2, nt = tile & 3;
  int lane = (int)(threadIdx.x & 31);
  int len = (KT + SK_ - 1) / SK_;
  int k0 = kc * len;
  int k1 = k0 + len; if (k1 > KT) k1 = KT;
  v8f c = {};
  for (int ks = k0; ks < k1; ++ks) {
    v16h a = frag_a_h(hsA, hsStride, mt * 16, ks * 32);
    v16h b = frag_b_pk(fcwT, 4, ks, nt);
    c = WMMA_F32(a, b, c);
  }
  float* dst = part + ((size_t)(kc * 20 + tile) << 8);
#pragma unroll
  for (int r = 0; r < 8; ++r)
    dst[(r + 8 * (lane >> 4)) * 16 + (lane & 15)] = c[r];
}

// ---- fixed-order reduction of all partials + biases -> out[16][64] (deterministic) ----
__global__ void k_reduce(const float* part, const float* b1, const float* b2, const float* b3,
                         float* out) {
  int t = blockIdx.x * blockDim.x + threadIdx.x;
  if (t >= B_ * OUT_) return;
  int b = t / OUT_, o = t % OUT_;
  int nt = o >> 4, n16 = o & 15;
  float acc = 5.0f * (b1[o] + b2[o] + b3[o]);
  for (int br = 0; br < 3; ++br)
    for (int kc = 0; kc < SK_; ++kc)
      for (int mt = 0; mt < 5; ++mt)
        acc += part[(((size_t)((br * SK_ + kc) * 5 + mt) * 4 + nt) << 8) + b * 16 + n16];
  out[t] = acc;
}

// ---------------- host ----------------
extern "C" void kernel_launch(void* const* d_in, const int* in_sizes, int n_in,
                              void* d_out, int out_size, void* d_ws, size_t ws_size,
                              hipStream_t stream) {
  (void)in_sizes; (void)n_in; (void)out_size; (void)ws_size;
  const float* board = (const float*)d_in[0];
  const float* order = (const float*)d_in[1];
  const float* msg   = (const float*)d_in[2];
  const float* adj   = (const float*)d_in[3];
  const float* g1w = (const float*)d_in[4];   const float* g1b = (const float*)d_in[5];
  const float* g2w = (const float*)d_in[6];   const float* g2b = (const float*)d_in[7];
  const float* lbw_i = (const float*)d_in[8]; const float* lbw_h = (const float*)d_in[9];
  const float* lbb_i = (const float*)d_in[10];const float* lbb_h = (const float*)d_in[11];
  const float* fb3w = (const float*)d_in[12]; const float* fb3b = (const float*)d_in[13];
  const float* fo1w = (const float*)d_in[14]; const float* fo1b = (const float*)d_in[15];
  const float* fo2w = (const float*)d_in[16]; const float* fo2b = (const float*)d_in[17];
  const float* low_i = (const float*)d_in[18];const float* low_h = (const float*)d_in[19];
  const float* lob_i = (const float*)d_in[20];const float* lob_h = (const float*)d_in[21];
  const float* fo3w = (const float*)d_in[22]; const float* fo3b = (const float*)d_in[23];
  const float* fm1w = (const float*)d_in[24]; const float* fm1b = (const float*)d_in[25];
  const float* fm2w = (const float*)d_in[26]; const float* fm2b = (const float*)d_in[27];
  const float* lmw_i = (const float*)d_in[28];const float* lmw_h = (const float*)d_in[29];
  const float* lmb_i = (const float*)d_in[30];const float* lmb_h = (const float*)d_in[31];
  const float* fm3w = (const float*)d_in[32]; const float* fm3b = (const float*)d_in[33];
  float* out = (float*)d_out;

  char* ws = (char*)d_ws;
  size_t off = 0;
  auto alloc = [&](size_t bytes) -> char* {
    char* p = ws + off;
    off += (bytes + 255) & ~(size_t)255;
    return p;
  };
  float* bf   = (float*)alloc((size_t)BPS_ * 162 * 4);
  float* ofF  = (float*)alloc((size_t)BPS_ * 32 * 4);
  float* mfF  = (float*)alloc((size_t)BPS_ * 162 * 4);
  float* gxB  = (float*)alloc((size_t)P_ * S_ * 656 * 16 * 4);
  float* gxM  = (float*)alloc((size_t)P_ * S_ * 656 * 16 * 4);
  float* gxO  = (float*)alloc((size_t)P_ * S_ * 128 * 16 * 4);
  _Float16* hsB = (_Float16*)alloc((size_t)80 * 32416 * 2);
  _Float16* hsM = (_Float16*)alloc((size_t)80 * 32416 * 2);
  _Float16* hsO = (_Float16*)alloc((size_t)80 * 6400 * 2);
  float* part = (float*)alloc((size_t)3 * SK_ * 20 * 256 * 4);
  _Float16* g1wT  = (_Float16*)alloc((size_t)32 * 32 * 2);
  _Float16* adjA  = (_Float16*)alloc((size_t)96 * 96 * 2);
  _Float16* fo1wT = (_Float16*)alloc((size_t)32 * 128 * 2);
  _Float16* fo2wT = (_Float16*)alloc((size_t)128 * 32 * 2);
  _Float16* fm1wT = (_Float16*)alloc((size_t)320 * 128 * 2);
  _Float16* fm2wT = (_Float16*)alloc((size_t)128 * 176 * 2);
  _Float16* wihB  = (_Float16*)alloc((size_t)192 * 656 * 2);
  _Float16* whhB  = (_Float16*)alloc((size_t)192 * 656 * 2);
  _Float16* wihM  = (_Float16*)alloc((size_t)192 * 656 * 2);
  _Float16* whhM  = (_Float16*)alloc((size_t)192 * 656 * 2);
  _Float16* wihO  = (_Float16*)alloc((size_t)32 * 128 * 2);
  _Float16* whhO  = (_Float16*)alloc((size_t)32 * 128 * 2);
  _Float16* fb3wT = (_Float16*)alloc((size_t)32416 * 64 * 2);
  _Float16* fm3wT = (_Float16*)alloc((size_t)32416 * 64 * 2);
  _Float16* fo3wT = (_Float16*)alloc((size_t)6400 * 64 * 2);

  struct Conv { const float* src; _Float16* dst; int N, K, Kpad, Npad; };
  Conv convs[] = {
    { g1w,   g1wT,  32, 32, 32, 32 },
    { fo1w,  fo1wT, 128, 32, 32, 128 },
    { fo2w,  fo2wT, 32, 128, 128, 32 },
    { fm1w,  fm1wT, 128, 320, 320, 128 },
    { fm2w,  fm2wT, 162, 128, 128, 176 },
    { lbw_i, wihB,  648, 162, 192, 656 },
    { lbw_h, whhB,  648, 162, 192, 656 },
    { lmw_i, wihM,  648, 162, 192, 656 },
    { lmw_h, whhM,  648, 162, 192, 656 },
    { low_i, wihO,  128, 32, 32, 128 },
    { low_h, whhO,  128, 32, 32, 128 },
    { fb3w,  fb3wT, 64, 32400, 32416, 64 },
    { fm3w,  fm3wT, 64, 32400, 32416, 64 },
    { fo3w,  fo3wT, 64, 6400, 6400, 64 },
  };
  for (int i = 0; i < (int)(sizeof(convs) / sizeof(convs[0])); ++i) {
    int total = convs[i].Kpad * convs[i].Npad;
    int grid = (total + 255) / 256; if (grid > 4096) grid = 4096;
    k_conv_wT_pk<<<grid, 256, 0, stream>>>(convs[i].src, convs[i].dst,
                                           convs[i].N, convs[i].K, convs[i].Kpad, convs[i].Npad);
  }
  k_conv_adj<<<(96 * 96 + 255) / 256, 256, 0, stream>>>(adj, adjA);

  // feature extraction
  k_gcn<<<BPS_, 128, 0, stream>>>(board, adj, g1wT, g1b, g2w, g2b, adjA, bf);
  k_of<<<BPS_ / 16, 32, 0, stream>>>(order, fo1wT, fo1b, fo2wT, fo2b, ofF);
  k_mf<<<BPS_ / 16, 128, 0, stream>>>(msg, fm1wT, fm1b, fm2wT, fm2b, mfF);

  // precompute x @ Wih^T + b_ih (parallel over all steps, stored transposed [blk][n][b])
  k_gx<<<P_ * S_, 128, 0, stream>>>(bf, 162, 192, wihB, lbb_i, 648, 41, gxB);
  k_gx<<<P_ * S_, 128, 0, stream>>>(mfF, 162, 192, wihM, lmb_i, 648, 41, gxM);
  k_gx<<<P_ * S_, 128, 0, stream>>>(ofF, 32, 32, wihO, lob_i, 128, 8, gxO);

  // sequential recurrences (persistent single workgroup each, state in LDS)
  size_t smB = (size_t)80 * 192 * 2 + (size_t)656 * 80 * 4 + (size_t)162 * 80 * 4; // ~286 KB
  size_t smO = (size_t)80 * 32 * 2 + (size_t)128 * 80 * 4 + (size_t)32 * 80 * 4;   // ~56 KB
  k_rec<<<1, 1024, smB, stream>>>(gxB, whhB, lbb_h, 162, 192, 41, hsB, 32416);
  k_rec<<<1, 1024, smB, stream>>>(gxM, whhM, lmb_h, 162, 192, 41, hsM, 32416);
  k_rec<<<1, 1024, smO, stream>>>(gxO, whhO, lob_h, 32, 32, 8, hsO, 6400);

  // final projections (split-K partials) + deterministic fixed-order reduction
  k_proj<<<SK_ * 20, 32, 0, stream>>>(hsB, 32416, 32416 / 32, fb3wT, part + 0 * SK_ * 20 * 256);
  k_proj<<<SK_ * 20, 32, 0, stream>>>(hsO, 6400, 6400 / 32, fo3wT, part + 1 * SK_ * 20 * 256);
  k_proj<<<SK_ * 20, 32, 0, stream>>>(hsM, 32416, 32416 / 32, fm3wT, part + 2 * SK_ * 20 * 256);
  k_reduce<<<4, 256, 0, stream>>>(part, fb3b, fo3b, fm3b, out);
}
